// TopPMoE_45148696217031
// MI455X (gfx1250) — matrix operand order repaired and checked by reading, not compiled, not run on hardware
//
#include <hip/hip_runtime.h>
#include <hip/hip_bf16.h>

// ---------------------------------------------------------------------------
// Problem constants (match reference)
// ---------------------------------------------------------------------------
#define Bb   4
#define Ss   1024
#define Hh   512
#define NHh  8
#define HDd  64
#define Ii   2048
#define Ee   8
#define MM   (Bb * Ss)      // 4096 token rows
#define H3   (3 * Hh)       // 1536

// ---------------------------------------------------------------------------
// WMMA / TDM types (CDNA5 / gfx1250, wave32)
// ---------------------------------------------------------------------------
typedef __attribute__((ext_vector_type(16))) __bf16        v16bf;
typedef __attribute__((ext_vector_type(8)))  float         v8f;
typedef __attribute__((ext_vector_type(4)))  unsigned int  v4u;
typedef __attribute__((ext_vector_type(4)))  unsigned int  u32x4;
typedef __attribute__((ext_vector_type(8)))  int           i32x8;
typedef __attribute__((ext_vector_type(4)))  int           i32x4;

union Frag16 {
  v16bf          v;
  unsigned short u[16];
  v4u            q[2];
};

__device__ __forceinline__ unsigned short f2bf(float f) {
  union { float f; unsigned int u; } c;
  c.f = f;
  unsigned int r = (c.u + 0x7FFFu + ((c.u >> 16) & 1u)) >> 16;  // RNE
  return (unsigned short)r;
}

__device__ __forceinline__ float geluf(float v) {
  return 0.5f * v * (1.0f + erff(v * 0.70710678118654752f));
}

__device__ __forceinline__ v8f wmma_bf16(v16bf a, v16bf b, v8f c) {
  // (neg_a, A, neg_b, B, c_mod, C, reuse_a, reuse_b)
  return __builtin_amdgcn_wmma_f32_16x16x32_bf16(false, a, false, b,
                                                 (short)0, c, false, false);
}

// A fragment: row-major A [*, lda] bf16, 16x32 tile at (m0, k0).
// 16-bit A layout (contiguous K, matching B): row = lane&15 for all lanes;
// lanes 0-15 carry K k0+{0..15}, lanes 16-31 carry K k0+{16..31}.
__device__ __forceinline__ v16bf frag_a(const unsigned short* __restrict__ A,
                                        int lda, int m0, int k0, int lane) {
  const int r16 = lane & 15, hi = lane >> 4;
  const unsigned short* p = A + (size_t)(m0 + r16) * lda + k0;
  Frag16 f;
  f.q[0] = *(const v4u*)(p + hi * 16);
  f.q[1] = *(const v4u*)(p + hi * 16 + 8);
  return f.v;
}

// B fragment from an N-major ("W-transposed") matrix in GLOBAL memory:
// B[k][n] = W[n0+n][k0+k], W row-major [N, ldw].
__device__ __forceinline__ v16bf frag_bT(const unsigned short* __restrict__ W,
                                         int ldw, int n0, int k0, int lane) {
  const int n = n0 + (lane & 15), hi = lane >> 4;
  const unsigned short* p = W + (size_t)n * ldw + k0 + hi * 16;
  Frag16 f;
  f.q[0] = *(const v4u*)(p);
  f.q[1] = *(const v4u*)(p + 8);
  return f.v;
}

// B fragment from the LDS-staged tile (row stride LDSB_STRIDE ushorts).
#define LDSB_STRIDE 40  // 32 K-elems (64B) + 16B TDM pad -> bank-conflict-free
__device__ __forceinline__ v16bf frag_b_lds(const unsigned short* __restrict__ T,
                                            int nLocal, int lane) {
  const int n = nLocal + (lane & 15), hi = lane >> 4;
  const unsigned short* p = T + n * LDSB_STRIDE + hi * 16;
  Frag16 f;
  f.q[0] = *(const v4u*)(p);
  f.q[1] = *(const v4u*)(p + 8);
  return f.v;
}

// ---------------------------------------------------------------------------
// Tensor Data Mover: stage a [tile 32(K) x 128(N)] bf16 tile of an N-major
// matrix (row stride = K elements) into LDS with 16B padding per 64B row.
// D# layout per CDNA5 ISA (group0 128b, group1 256b).
// This toolchain exposes the 6-arg builtin:
//   (u32x4 g0, i32x8 g1, i32x4 g2, i32x4 g3, i32x8 g4, i32 cpol)
// ---------------------------------------------------------------------------
__device__ __forceinline__ void tdm_load_btile(const unsigned short* gsrc,
                                               int K, int N, unsigned lds_off) {
  const unsigned long long ga = (unsigned long long)(size_t)gsrc;
  u32x4 g0;
  g0.x = 1u;                                           // count=1, user desc
  g0.y = lds_off;                                      // lds_addr
  g0.z = (unsigned)(ga & 0xFFFFFFFFull);               // global_addr[31:0]
  g0.w = (unsigned)((ga >> 32) & 0x1FFFFFFull) | (2u << 30);  // [56:32] | type=2
  i32x8 g1;
  // data_size=1(2B) | pad_enable | pad_interval=3(64B) | pad_amount=3(4 dw)
  g1[0] = (int)((1u << 16) | (1u << 20) | (3u << 22) | (3u << 25));
  g1[1] = (int)(((unsigned)K & 0xFFFFu) << 16);        // tensor_dim0[15:0]
  g1[2] = (int)((((unsigned)K >> 16) & 0xFFFFu) |
                (((unsigned)N & 0xFFFFu) << 16));      // dim0 hi | dim1 lo
  g1[3] = (int)((((unsigned)N >> 16) & 0xFFFFu) |
                (32u << 16));                          // dim1 hi | tile_dim0=32
  g1[4] = (int)128u;                                   // tile_dim1=128, tile_dim2=0
  g1[5] = (int)(unsigned)K;                            // tensor_dim0_stride lo32
  g1[6] = 0;                                           // stride0 hi | stride1 lo
  g1[7] = 0;
  const i32x4 gz4 = {0, 0, 0, 0};
  const i32x8 gz8 = {0, 0, 0, 0, 0, 0, 0, 0};
  __builtin_amdgcn_tensor_load_to_lds(g0, g1, gz4, gz4, gz8, 0);
}

// ---------------------------------------------------------------------------
// Elementwise conversion kernels
// ---------------------------------------------------------------------------
__global__ __launch_bounds__(256) void cvt_kernel(const float* __restrict__ in,
                                                  unsigned short* __restrict__ out,
                                                  long n) {
  long i = (long)blockIdx.x * blockDim.x + threadIdx.x;
  long stride = (long)gridDim.x * blockDim.x;
  for (; i < n; i += stride) out[i] = f2bf(in[i]);
}

// in [E, R, C] f32 -> out [E, C, R] bf16 (per-expert transpose)
__global__ __launch_bounds__(256) void cvtT_kernel(const float* __restrict__ in,
                                                   unsigned short* __restrict__ out,
                                                   int Emats, int R, int C) {
  long n = (long)Emats * R * C;
  long i = (long)blockIdx.x * blockDim.x + threadIdx.x;
  long stride = (long)gridDim.x * blockDim.x;
  long RC = (long)R * C;
  for (; i < n; i += stride) {
    long e = i / RC, rem = i - e * RC;
    long r = rem / C, c = rem - r * C;
    out[e * RC + c * (long)R + r] = f2bf(in[i]);
  }
}

__global__ __launch_bounds__(256) void zero_kernel(float* __restrict__ p, long n) {
  long i = (long)blockIdx.x * blockDim.x + threadIdx.x;
  long stride = (long)gridDim.x * blockDim.x;
  for (; i < n; i += stride) p[i] = 0.0f;
}

// ---------------------------------------------------------------------------
// WMMA GEMM with TDM-staged B tiles.
// C[M,N] = A[M,K] (bf16 row-major) x B, B[k][n] stored N-major as Bw[n][k].
// Block: 256 thr = 8 waves as 4(m) x 2(n); block tile 128x128; wave tile 32x64.
// B tile [128 n x 32 k] double-buffered in LDS via tensor_load_to_lds.
// Epilogue by MODE:
//   0: bf16 out = acc + bias[n]
//   1: f32  out = acc + bias[n]
//   2: bf16 out = gelu(acc + bias[n])
//   3: f32  accum[m*N+n] += (acc + bias[n]) * gates[m*E + expert]
// ---------------------------------------------------------------------------
template <int MODE>
__global__ __launch_bounds__(256) void gemm_wmma_kernel(
    const unsigned short* __restrict__ A, const unsigned short* __restrict__ Bw,
    const float* __restrict__ bias, unsigned short* __restrict__ outB,
    float* __restrict__ outF, const float* __restrict__ gates, int expert,
    int M, int N, int K) {
  __shared__ __align__(16) unsigned short ldsB[2][128 * LDSB_STRIDE];  // 20 KB

  const int wave = threadIdx.x >> 5;
  const int lane = threadIdx.x & 31;
  const int r16 = lane & 15, hi = lane >> 4;
  const int mw = wave & 3, nw = wave >> 2;
  const int nBlk = blockIdx.x * 128;
  const int m0 = blockIdx.y * 128 + mw * 32;
  (void)M;

  const v8f vz = {0.f, 0.f, 0.f, 0.f, 0.f, 0.f, 0.f, 0.f};
  v8f acc[2][4];
#pragma unroll
  for (int mt = 0; mt < 2; mt++)
#pragma unroll
    for (int t = 0; t < 4; t++) acc[mt][t] = vz;

  const unsigned short* Bblk = Bw + (size_t)nBlk * K;
  const int kTiles = K / 32;

  if (wave == 0) {
    tdm_load_btile(Bblk, K, N, (unsigned)(size_t)&ldsB[0][0]);
    __builtin_amdgcn_s_wait_tensorcnt(0);
  }
  __syncthreads();

  for (int kt = 0; kt < kTiles; kt++) {
    const int cur = kt & 1, nxt = cur ^ 1;
    if (wave == 0 && kt + 1 < kTiles)
      tdm_load_btile(Bblk + (size_t)(kt + 1) * 32, K, N,
                     (unsigned)(size_t)&ldsB[nxt][0]);

    const int k0 = kt * 32;
    __builtin_prefetch(A + (size_t)(m0 + r16) * K + k0 + 64, 0, 0);
    v16bf a0 = frag_a(A, K, m0, k0, lane);
    v16bf a1 = frag_a(A, K, m0 + 16, k0, lane);
#pragma unroll
    for (int t = 0; t < 4; t++) {
      v16bf b = frag_b_lds(&ldsB[cur][0], nw * 64 + t * 16, lane);
      acc[0][t] = wmma_bf16(a0, b, acc[0][t]);
      acc[1][t] = wmma_bf16(a1, b, acc[1][t]);
    }

    if (wave == 0) __builtin_amdgcn_s_wait_tensorcnt(0);
    __syncthreads();
  }

#pragma unroll
  for (int t = 0; t < 4; t++) {
    const int n = nBlk + nw * 64 + t * 16 + r16;
    const float bs = bias[n];
#pragma unroll
    for (int mt = 0; mt < 2; mt++) {
#pragma unroll
      for (int r = 0; r < 8; r++) {
        const int m = m0 + mt * 16 + r + hi * 8;  // C-layout: M = vgpr + 8*hi
        const float v = acc[mt][t][r] + bs;
        if (MODE == 0) {
          outB[(size_t)m * N + n] = f2bf(v);
        } else if (MODE == 1) {
          outF[(size_t)m * N + n] = v;
        } else if (MODE == 2) {
          outB[(size_t)m * N + n] = f2bf(geluf(v));
        } else {
          const float g = gates[(size_t)m * Ee + expert];
          outF[(size_t)m * N + n] += v * g;  // unique writer per element
        }
      }
    }
  }
}

// ---------------------------------------------------------------------------
// Attention: one block per (b, head, 16-row q block).  qkvb is bf16 [MM, 3H]
// (q | k | v concatenated per row).  Output: ctxb bf16 [MM, H].
// ---------------------------------------------------------------------------
__global__ __launch_bounds__(256) void attn_kernel(
    const unsigned short* __restrict__ qkvb, unsigned short* __restrict__ ctxb) {
  __shared__ float sc[16 * Ss];  // 64 KB score/attn tile

  const int blk = blockIdx.x;           // b*NH*(S/16) + h*(S/16) + qb
  const int qb = blk & (Ss / 16 - 1);
  const int h = (blk >> 6) & (NHh - 1);
  const int b = blk >> 9;
  const int wave = threadIdx.x >> 5;
  const int lane = threadIdx.x & 31;
  const int r16 = lane & 15, hi = lane >> 4;

  const size_t rowBase = (size_t)b * Ss;
  const unsigned short* Qp = qkvb + rowBase * H3 + (size_t)h * HDd;
  const unsigned short* Kp = Qp + Hh;
  const unsigned short* Vp = Qp + 2 * Hh;
  const int m0 = qb * 16;

  // ---- phase 1: scores = (Q K^T) / sqrt(HD) -> LDS ----
  for (int kb = wave; kb < Ss / 16; kb += 8) {
    v8f acc = {0.f, 0.f, 0.f, 0.f, 0.f, 0.f, 0.f, 0.f};
#pragma unroll
    for (int kk = 0; kk < HDd; kk += 32) {
      v16bf a = frag_a(Qp, H3, m0, kk, lane);
      v16bf kf = frag_bT(Kp, H3, kb * 16, kk, lane);  // B[k][n] = K[key n][feat k]
      acc = wmma_bf16(a, kf, acc);
    }
#pragma unroll
    for (int r = 0; r < 8; r++)
      sc[(r + hi * 8) * Ss + kb * 16 + r16] = acc[r] * 0.125f;
  }
  __syncthreads();

  // ---- phase 2: row softmax (wave handles rows 2w, 2w+1) ----
#pragma unroll
  for (int rr = 0; rr < 2; rr++) {
    float* rowp = sc + (wave * 2 + rr) * Ss;
    float mx = -3.0e38f;
    for (int j = lane; j < Ss; j += 32) mx = fmaxf(mx, rowp[j]);
#pragma unroll
    for (int off = 16; off; off >>= 1) mx = fmaxf(mx, __shfl_xor(mx, off));
    float sum = 0.f;
    for (int j = lane; j < Ss; j += 32) {
      float e = __expf(rowp[j] - mx);
      rowp[j] = e;
      sum += e;
    }
#pragma unroll
    for (int off = 16; off; off >>= 1) sum += __shfl_xor(sum, off);
    const float inv = 1.f / sum;
    for (int j = lane; j < Ss; j += 32) rowp[j] *= inv;
  }
  __syncthreads();

  // ---- phase 3: ctx = attn @ V, K-dim split across waves ----
  const v8f vz = {0.f, 0.f, 0.f, 0.f, 0.f, 0.f, 0.f, 0.f};
  v8f ctx[4];
#pragma unroll
  for (int t = 0; t < 4; t++) ctx[t] = vz;

  for (int kb = wave * 4; kb < wave * 4 + 4; kb++) {
    const int k0 = kb * 32;
    Frag16 a;
#pragma unroll
    for (int e2 = 0; e2 < 8; e2++)
      a.u[e2] = f2bf(sc[r16 * Ss + k0 + hi * 16 + e2]);
#pragma unroll
    for (int e2 = 0; e2 < 8; e2++)
      a.u[8 + e2] = f2bf(sc[r16 * Ss + k0 + hi * 16 + 8 + e2]);
#pragma unroll
    for (int t = 0; t < 4; t++) {
      Frag16 bf;
#pragma unroll
      for (int e2 = 0; e2 < 16; e2++)
        bf.u[e2] = Vp[(size_t)(k0 + hi * 16 + e2) * H3 + t * 16 + r16];
      ctx[t] = wmma_bf16(a.v, bf.v, ctx[t]);
    }
  }
  __syncthreads();  // scores consumed; reuse LDS for reduction

  float* red = sc;  // [16][64]
  for (int i = threadIdx.x; i < 16 * HDd; i += 256) red[i] = 0.f;
  __syncthreads();
#pragma unroll
  for (int t = 0; t < 4; t++)
#pragma unroll
    for (int r = 0; r < 8; r++)
      atomicAdd(&red[(r + hi * 8) * HDd + t * 16 + r16], ctx[t][r]);
  __syncthreads();
  for (int i = threadIdx.x; i < 16 * HDd; i += 256) {
    const int r = i >> 6, c = i & 63;
    ctxb[(rowBase + m0 + r) * Hh + (size_t)h * HDd + c] = f2bf(red[i]);
  }
}

// ---------------------------------------------------------------------------
// LayerNorm: out = LN(xin + addin) * g + b.  One wave per row of Hh.
// ---------------------------------------------------------------------------
__global__ __launch_bounds__(256) void ln_kernel(
    const float* __restrict__ xin, const float* __restrict__ addin,
    const float* __restrict__ g, const float* __restrict__ bvec,
    float* __restrict__ outF, unsigned short* __restrict__ outB) {
  const int wave = threadIdx.x >> 5, lane = threadIdx.x & 31;
  const size_t m = (size_t)blockIdx.x * 8 + wave;
  const float* xr = xin + m * Hh;
  const float* ar = addin + m * Hh;
  float s = 0.f, s2 = 0.f;
  for (int j = lane; j < Hh; j += 32) {
    const float v = xr[j] + ar[j];
    s += v;
    s2 += v * v;
  }
#pragma unroll
  for (int off = 16; off; off >>= 1) {
    s += __shfl_xor(s, off);
    s2 += __shfl_xor(s2, off);
  }
  const float mean = s * (1.0f / Hh);
  const float var = s2 * (1.0f / Hh) - mean * mean;
  const float inv = rsqrtf(var + 1e-12f);
  for (int j = lane; j < Hh; j += 32) {
    const float v = (xr[j] + ar[j] - mean) * inv * g[j] + bvec[j];
    outF[m * Hh + j] = v;
    if (outB) outB[m * Hh + j] = f2bf(v);
  }
}

// ---------------------------------------------------------------------------
// Top-p gating: one wave per token row.
// ---------------------------------------------------------------------------
__global__ __launch_bounds__(256) void gate_kernel(
    const float* __restrict__ x1, const float* __restrict__ gw,
    float* __restrict__ gates, float* __restrict__ lbsum) {
  const int wave = threadIdx.x >> 5, lane = threadIdx.x & 31;
  const size_t m = (size_t)blockIdx.x * 8 + wave;
  const float* xr = x1 + m * Hh;
  float acc[Ee];
#pragma unroll
  for (int e = 0; e < Ee; e++) acc[e] = 0.f;
  for (int j = lane; j < Hh; j += 32) {
    const float xv = xr[j];
#pragma unroll
    for (int e = 0; e < Ee; e++) acc[e] += xv * gw[e * Hh + j];
  }
#pragma unroll
  for (int e = 0; e < Ee; e++)
#pragma unroll
    for (int off = 16; off; off >>= 1) acc[e] += __shfl_xor(acc[e], off);

  if (lane == 0) {
    float mx = acc[0];
#pragma unroll
    for (int e = 1; e < Ee; e++) mx = fmaxf(mx, acc[e]);
    float g[Ee], s = 0.f;
#pragma unroll
    for (int e = 0; e < Ee; e++) {
      g[e] = __expf(acc[e] - mx);
      s += g[e];
    }
    const float invs0 = 1.f / s;
#pragma unroll
    for (int e = 0; e < Ee; e++) g[e] *= invs0;
    int idx[Ee];
#pragma unroll
    for (int e = 0; e < Ee; e++) idx[e] = e;
    for (int i = 0; i < Ee; i++)
      for (int j = i + 1; j < Ee; j++)
        if (g[idx[j]] > g[idx[i]]) {
          int t = idx[i];
          idx[i] = idx[j];
          idx[j] = t;
        }
    float cum = 0.f, keep[Ee];
    for (int i = 0; i < Ee; i++) {
      cum += g[idx[i]];
      keep[idx[i]] = (cum < 0.5f) ? 1.f : 0.f;
    }
    float s2 = 0.f;
#pragma unroll
    for (int e = 0; e < Ee; e++) {
      g[e] *= keep[e];
      s2 += g[e];
    }
    const float invs = 1.f / (s2 + 1e-10f);
#pragma unroll
    for (int e = 0; e < Ee; e++) {
      const float gv = g[e] * invs;
      gates[m * Ee + e] = gv;
      atomicAdd(&lbsum[e], gv);
    }
  }
}

__global__ void lb_kernel(const float* __restrict__ lbsum, float* __restrict__ out) {
  if (threadIdx.x == 0 && blockIdx.x == 0) {
    float acc = 0.f;
#pragma unroll
    for (int e = 0; e < Ee; e++) {
      const float d = lbsum[e] / (float)MM - (1.0f / Ee);
      acc += d * d;
    }
    out[0] = acc / (float)Ee;
  }
}

// ---------------------------------------------------------------------------
// Host launcher
// ---------------------------------------------------------------------------
extern "C" void kernel_launch(void* const* d_in, const int* in_sizes, int n_in,
                              void* d_out, int out_size, void* d_ws,
                              size_t ws_size, hipStream_t stream) {
  (void)in_sizes; (void)n_in; (void)out_size; (void)ws_size;

  const float* x      = (const float*)d_in[0];
  const float* in_w   = (const float*)d_in[1];
  const float* in_b   = (const float*)d_in[2];
  const float* out_w  = (const float*)d_in[3];
  const float* out_b  = (const float*)d_in[4];
  const float* ln1_g  = (const float*)d_in[5];
  const float* ln1_b  = (const float*)d_in[6];
  const float* gate_w = (const float*)d_in[7];
  const float* w1     = (const float*)d_in[8];
  const float* b1     = (const float*)d_in[9];
  const float* w2     = (const float*)d_in[10];
  const float* b2     = (const float*)d_in[11];
  const float* ln2_g  = (const float*)d_in[12];
  const float* ln2_b  = (const float*)d_in[13];
  float* out = (float*)d_out;

  char* w = (char*)d_ws;
  auto alloc = [&](size_t bytes) -> char* {
    char* p = w;
    w += (bytes + 255) & ~(size_t)255;
    return p;
  };
  unsigned short* xb    = (unsigned short*)alloc((size_t)MM * Hh * 2);
  unsigned short* wqkvb = (unsigned short*)alloc((size_t)H3 * Hh * 2);
  unsigned short* woutb = (unsigned short*)alloc((size_t)Hh * Hh * 2);
  unsigned short* w1T   = (unsigned short*)alloc((size_t)Ee * Ii * Hh * 2);
  unsigned short* w2T   = (unsigned short*)alloc((size_t)Ee * Hh * Ii * 2);
  unsigned short* qkvb  = (unsigned short*)alloc((size_t)MM * H3 * 2);
  unsigned short* ctxb  = (unsigned short*)alloc((size_t)MM * Hh * 2);
  float* attn_out       = (float*)alloc((size_t)MM * Hh * 4);
  float* x1             = (float*)alloc((size_t)MM * Hh * 4);
  unsigned short* x1b   = (unsigned short*)alloc((size_t)MM * Hh * 2);
  float* gates          = (float*)alloc((size_t)MM * Ee * 4);
  unsigned short* hbuf  = (unsigned short*)alloc((size_t)MM * Ii * 2);
  float* ffn            = (float*)alloc((size_t)MM * Hh * 4);
  float* lbsum          = (float*)alloc(64 * 4);

  const dim3 blk(256);

  // init accumulators
  zero_kernel<<<2048, blk, 0, stream>>>(ffn, (long)MM * Hh);
  zero_kernel<<<1, blk, 0, stream>>>(lbsum, 64);

  // f32 -> bf16 conversions (weights transposed per-expert for contiguous-K B)
  cvt_kernel<<<2048, blk, 0, stream>>>(x, xb, (long)MM * Hh);
  cvt_kernel<<<2048, blk, 0, stream>>>(in_w, wqkvb, (long)H3 * Hh);
  cvt_kernel<<<1024, blk, 0, stream>>>(out_w, woutb, (long)Hh * Hh);
  cvtT_kernel<<<4096, blk, 0, stream>>>(w1, w1T, Ee, Hh, Ii);  // [E,H,I]->[E,I,H]
  cvtT_kernel<<<4096, blk, 0, stream>>>(w2, w2T, Ee, Ii, Hh);  // [E,I,H]->[E,H,I]

  // qkv = x @ in_proj_w^T + in_proj_b  (bf16 out)
  gemm_wmma_kernel<0><<<dim3(H3 / 128, MM / 128), blk, 0, stream>>>(
      xb, wqkvb, in_b, qkvb, nullptr, nullptr, 0, MM, H3, Hh);

  // attention -> ctxb (bf16 [MM, H], head-interleaved)
  attn_kernel<<<Bb * NHh * (Ss / 16), blk, 0, stream>>>(qkvb, ctxb);

  // attn_out = ctx @ out_proj_w^T + out_proj_b  (f32 out)
  gemm_wmma_kernel<1><<<dim3(Hh / 128, MM / 128), blk, 0, stream>>>(
      ctxb, woutb, out_b, nullptr, attn_out, nullptr, 0, MM, Hh, Hh);

  // x1 = LN(x + attn_out); also bf16 copy for the expert GEMMs
  ln_kernel<<<MM / 8, blk, 0, stream>>>(x, attn_out, ln1_g, ln1_b, x1, x1b);

  // top-p gates + LB-loss partial sums
  gate_kernel<<<MM / 8, blk, 0, stream>>>(x1, gate_w, gates, lbsum);

  // dense expert stack, one expert at a time (hbuf reused)
  for (int e = 0; e < Ee; e++) {
    gemm_wmma_kernel<2><<<dim3(Ii / 128, MM / 128), blk, 0, stream>>>(
        x1b, w1T + (size_t)e * Ii * Hh, b1 + (size_t)e * Ii, hbuf, nullptr,
        nullptr, 0, MM, Ii, Hh);
    gemm_wmma_kernel<3><<<dim3(Hh / 128, MM / 128), blk, 0, stream>>>(
        hbuf, w2T + (size_t)e * Hh * Ii, b2 + (size_t)e * Hh, nullptr, ffn,
        gates, e, MM, Hh, Ii);
  }

  // out = LN(x1 + ffn)
  ln_kernel<<<MM / 8, blk, 0, stream>>>(x1, ffn, ln2_g, ln2_b, out, nullptr);

  // lb_loss scalar
  lb_kernel<<<1, 32, 0, stream>>>(lbsum, out + (size_t)MM * Hh);
}
